// SimpleRNN_64682207478504
// MI455X (gfx1250) — matrix operand order repaired and checked by reading, not compiled, not run on hardware
//
#include <hip/hip_runtime.h>
#include <hip/hip_bf16.h>
#include <math.h>

// ---------------------------------------------------------------------------
// SimpleRNN for MI455X (gfx1250), wave32 + WMMA bf16.
//   Phase 1: inp[b,s,h] = sum_i X[b,s,i] * W[h,i]    (parallel WMMA GEMM)
//            written directly into d_out (B,S,H layout).
//   Phase 2: persistent 16-block kernel, U slice resident in LDS per WGP,
//            1024 sequential steps with a global release/acquire barrier.
// ---------------------------------------------------------------------------

typedef __attribute__((ext_vector_type(16))) __bf16 v16bf;
typedef __attribute__((ext_vector_type(8)))  float  v8f;

#define BATCH   64
#define SEQ     1024
#define IN_DIM  512
#define HID     1024
#define NBLK    16            // persistent blocks (one per WGP)
#define LDU     (HID + 8)     // padded LDS row stride (bf16 elems) to dodge bank conflicts

// ---- WMMA fragment helpers (layouts per CDNA5 ISA 7.12.2) ------------------

// A-matrix 16x32 bf16 from an fp32 row-major source (ld in floats).
// lane<16: elems 0..7 = K 0..7, 8..15 = K 16..23 ; lane>=16: K 8..15 / 24..31.
__device__ __forceinline__ v16bf load_a_f32(const float* __restrict__ base, int ld,
                                            int m, int k, int lane) {
    int K0 = (lane < 16) ? 0 : 8;
    const float* p = base + (size_t)(m + (lane & 15)) * ld + k + K0;
    float4 c0 = *(const float4*)(p + 0);
    float4 c1 = *(const float4*)(p + 4);
    float4 c2 = *(const float4*)(p + 16);
    float4 c3 = *(const float4*)(p + 20);
    float f[16] = {c0.x, c0.y, c0.z, c0.w, c1.x, c1.y, c1.z, c1.w,
                   c2.x, c2.y, c2.z, c2.w, c3.x, c3.y, c3.z, c3.w};
    v16bf a;
#pragma unroll
    for (int i = 0; i < 16; ++i) a[i] = (__bf16)f[i];
    return a;
}

// A-matrix 16x32 bf16 from a bf16 row-major source (ld in elems).
__device__ __forceinline__ v16bf load_a_bf16(const __bf16* __restrict__ base, int ld,
                                             int m, int k, int lane) {
    int K0 = (lane < 16) ? 0 : 8;
    const __bf16* p = base + (size_t)(m + (lane & 15)) * ld + k + K0;
    union { v16bf v; uint4 q[2]; } u;
    u.q[0] = *(const uint4*)(p);        // 8 bf16 = 16B
    u.q[1] = *(const uint4*)(p + 16);   // 8 bf16 = 16B
    return u.v;
}

// B-matrix 32x16 bf16: B[k][n] = Src[n][k] (Src row-major, ld in elems).
// lane n = lane&15 ; lane<16: elems 0..7 = K 0..7, 8..15 = K 16..23 ;
// lane>=16: K 8..15 / 24..31 (same order as the A loaders).
__device__ __forceinline__ v16bf load_b_bf16(const __bf16* __restrict__ base, int ld,
                                             int n, int k, int lane) {
    int K0 = (lane < 16) ? 0 : 8;
    const __bf16* p = base + (size_t)(n + (lane & 15)) * ld + k + K0;
    union { v16bf v; uint4 q[2]; } u;
    u.q[0] = *(const uint4*)(p);
    u.q[1] = *(const uint4*)(p + 16);
    return u.v;
}

__device__ __forceinline__ v8f wmma_bf16(v16bf a, v16bf b, v8f c) {
    return __builtin_amdgcn_wmma_f32_16x16x32_bf16(false, a, false, b,
                                                   (short)0, c, false, false);
}

// ---- Phase 0: precision conversion (W -> bf16 ws, h0 -> bf16 ping buffer) --

__global__ __launch_bounds__(256) void convert_kernel(const float* __restrict__ W,
                                                      const float* __restrict__ h0,
                                                      __bf16* __restrict__ Wb,
                                                      __bf16* __restrict__ hb0) {
    int i = blockIdx.x * 256 + threadIdx.x;
    if (i < HID * IN_DIM) Wb[i] = (__bf16)W[i];
    if (i < BATCH * HID)  hb0[i] = (__bf16)h0[i];
}

// ---- Phase 1: inp = X @ W^T  (M=B*S=65536, N=HID, K=IN_DIM) ----------------
// 64x64 tile per block, 8 waves, each wave: 1 m-tile x 2 n-tiles.

__global__ __launch_bounds__(256) void gemm_inp_kernel(const float* __restrict__ X,
                                                       const __bf16* __restrict__ Wb,
                                                       float* __restrict__ out) {
    const int lane = threadIdx.x & 31;
    const int wave = threadIdx.x >> 5;
    const int mrow = blockIdx.x * 64 + (wave & 3) * 16;
    const int n0   = blockIdx.y * 64 + (wave >> 2) * 32;
    const int n1   = n0 + 16;

    v8f acc0 = {}, acc1 = {};
#pragma unroll 4
    for (int k = 0; k < IN_DIM; k += 32) {
        v16bf a  = load_a_f32 (X,  IN_DIM, mrow, k, lane);
        v16bf b0 = load_b_bf16(Wb, IN_DIM, n0,   k, lane);
        v16bf b1 = load_b_bf16(Wb, IN_DIM, n1,   k, lane);
        acc0 = wmma_bf16(a, b0, acc0);
        acc1 = wmma_bf16(a, b1, acc1);
    }
    const int nloc = lane & 15;
    const int moff = (lane < 16) ? 0 : 8;
#pragma unroll
    for (int g = 0; g < 8; ++g) {
        size_t r = (size_t)(mrow + g + moff) * HID;
        out[r + n0 + nloc] = acc0[g];
        out[r + n1 + nloc] = acc1[g];
    }
}

// ---- Phase 2: persistent recurrence --------------------------------------
// Block j owns output columns [64j, 64j+64); its U rows live in LDS (bf16)
// for all SEQ steps. h state ping-pongs between two bf16 buffers in L2.

__global__ __launch_bounds__(256) void rnn_recurrence_kernel(
    const float* __restrict__ U, float* __restrict__ out,
    __bf16* __restrict__ hbuf0, __bf16* __restrict__ hbuf1,
    unsigned int* __restrict__ ctr) {

    extern __shared__ __bf16 Ulds[];          // 64 rows x LDU bf16 (~129 KB)
    const int tid   = threadIdx.x;
    const int lane  = tid & 31;
    const int wave  = tid >> 5;
    const int nBase = blockIdx.x * 64;

    // Stage this block's U slice into LDS (fp32 -> bf16), 256 floats/thread.
    for (int idx = tid; idx < 64 * (HID / 4); idx += 256) {
        int rrow = idx >> 8;                  // / (HID/4)
        int ccol = (idx & 255) * 4;
        float4 v = *(const float4*)&U[(size_t)(nBase + rrow) * HID + ccol];
        __bf16* dst = &Ulds[rrow * LDU + ccol];
        dst[0] = (__bf16)v.x; dst[1] = (__bf16)v.y;
        dst[2] = (__bf16)v.z; dst[3] = (__bf16)v.w;
    }
    __syncthreads();

    const int mrow = (wave & 3) * 16;         // batch rows of this wave's tiles
    const int nl0  = (wave >> 2) * 32;        // local col tiles
    const int nl1  = nl0 + 16;
    const int nloc = lane & 15;
    const int moff = (lane < 16) ? 0 : 8;

    for (int s = 0; s < SEQ; ++s) {
        const __bf16* hprev = (s & 1) ? hbuf1 : hbuf0;
        __bf16*       hnext = (s & 1) ? hbuf0 : hbuf1;

        v8f acc0 = {}, acc1 = {};
#pragma unroll 4
        for (int k = 0; k < HID; k += 32) {
            v16bf a  = load_a_bf16(hprev, HID, mrow, k, lane);
            v16bf b0 = load_b_bf16(Ulds,  LDU, nl0,  k, lane);
            v16bf b1 = load_b_bf16(Ulds,  LDU, nl1,  k, lane);
            acc0 = wmma_bf16(a, b0, acc0);
            acc1 = wmma_bf16(a, b1, acc1);
        }

        // Fused epilogue: + inp, tanh, fp32 out, bf16 next-state (and h_final).
#pragma unroll
        for (int g = 0; g < 8; ++g) {
            int b = mrow + g + moff;
            {
                int h = nBase + nl0 + nloc;
                size_t oidx = ((size_t)b * SEQ + s) * HID + h;
                float v = tanhf(acc0[g] + out[oidx]);
                out[oidx] = v;
                hnext[b * HID + h] = (__bf16)v;
                if (s == SEQ - 1) out[(size_t)BATCH * SEQ * HID + b * HID + h] = v;
            }
            {
                int h = nBase + nl1 + nloc;
                size_t oidx = ((size_t)b * SEQ + s) * HID + h;
                float v = tanhf(acc1[g] + out[oidx]);
                out[oidx] = v;
                hnext[b * HID + h] = (__bf16)v;
                if (s == SEQ - 1) out[(size_t)BATCH * SEQ * HID + b * HID + h] = v;
            }
        }

        // Grid-wide step barrier (16 co-resident blocks).
        __syncthreads();
        if (tid == 0) {
            __threadfence();  // make hnext writes device-visible
            __hip_atomic_fetch_add(ctr, 1u, __ATOMIC_ACQ_REL, __HIP_MEMORY_SCOPE_AGENT);
        }
        const unsigned int target = (unsigned int)(NBLK * (s + 1));
        while (__hip_atomic_load(ctr, __ATOMIC_ACQUIRE, __HIP_MEMORY_SCOPE_AGENT) < target) {
            __builtin_amdgcn_s_sleep(2);
        }
        __threadfence();
    }
}

// ---------------------------------------------------------------------------

extern "C" void kernel_launch(void* const* d_in, const int* in_sizes, int n_in,
                              void* d_out, int out_size, void* d_ws, size_t ws_size,
                              hipStream_t stream) {
    (void)in_sizes; (void)n_in; (void)out_size; (void)ws_size;
    const float* X  = (const float*)d_in[0];   // (B,S,I)
    const float* h0 = (const float*)d_in[1];   // (B,H)
    const float* U  = (const float*)d_in[2];   // (H,H)
    const float* W  = (const float*)d_in[3];   // (H,I)
    float* out = (float*)d_out;                // (B,S,H) then (B,H) tail

    char* ws = (char*)d_ws;
    __bf16* Wb  = (__bf16*)ws;                                  // 1 MB
    __bf16* hb0 = (__bf16*)(ws + (1u << 20));                   // 128 KB
    __bf16* hb1 = (__bf16*)(ws + (1u << 20) + BATCH * HID * 2); // 128 KB
    unsigned int* ctr = (unsigned int*)(ws + (1u << 20) + 2 * BATCH * HID * 2);

    hipMemsetAsync(ctr, 0, 128, stream);
    convert_kernel<<<(HID * IN_DIM + 255) / 256, 256, 0, stream>>>(W, h0, Wb, hb0);
    gemm_inp_kernel<<<dim3((BATCH * SEQ) / 64, HID / 64), 256, 0, stream>>>(X, Wb, out);
    rnn_recurrence_kernel<<<NBLK, 256, 64 * LDU * sizeof(__bf16), stream>>>(U, out, hb0, hb1, ctr);
}
